// SS2D_89361089560832
// MI455X (gfx1250) — compile-verified, matrix operand order checked by reading, not run on hardware
//
#include <hip/hip_runtime.h>
#include <hip/hip_bf16.h>

// ---------------- problem constants ----------------
#define BB   2
#define HH   56
#define WW   56
#define LL   (HH*WW)      // 3136
#define DM   96           // d_model
#define DI   192          // d_inner
#define NS   16           // d_state
#define DTR  6            // dt_rank
#define KK   4            // directions
#define XPN  38           // dt_rank + 2*d_state
#define XPNP 48           // padded to multiple of 16

// chunked scan decomposition
#define NC    28                  // chunks per sequence
#define CS    (LL/NC)             // 112 steps per chunk
#define NCH   (BB*KK*DI)          // 1536 channels
#define ELEMS (NCH*NS)            // 24576 (channel, state) elements

typedef __attribute__((ext_vector_type(16))) _Float16 v16h;
typedef __attribute__((ext_vector_type(8)))  float    v8f;

// ---------------- scan-direction index math ----------------
// Index into x1cl (B, L, DI) giving xs[b,k,:,l] for direction k, seq pos l.
__device__ __forceinline__ int xs_src(int k, int l) {
    int ll = (k & 2) ? (LL - 1 - l) : l;          // k=2,3 are reversed sequences
    if (!(k & 1)) return ll;                      // k=0,2: h-scan order
    int i = ll / HH, j = ll - i * HH;             // k=1,3: v-scan order (W,H)
    int col = (j & 1) ? (WW - 1 - i) : i;
    return j * WW + col;
}

__device__ __forceinline__ float softplus_f(float x) {
    return (x > 20.f) ? x : __logf(1.f + __expf(x));
}

// ---------------- WMMA tile loaders (f32 -> f16) ----------------
// ISA 7.12.2 16-bit A-matrix 16x32 layout: lanes 0-15 row M=lane, halves 0..7 -> K 0..7,
// halves 8..15 -> K 16..23; lanes 16-31: K base +8.
__device__ __forceinline__ v16h load_tile16(const float* __restrict__ base,
                                            long row, int ld, int k0, int lane) {
    v16h v;
    const float* p = base + row * (long)ld + k0 + ((lane >> 4) << 3);
#pragma unroll
    for (int i = 0; i < 8; ++i) v[i]     = (_Float16)p[i];
#pragma unroll
    for (int i = 0; i < 8; ++i) v[i + 8] = (_Float16)p[i + 16];
    return v;
}

__device__ __forceinline__ v16h load_tile16_guard(const float* __restrict__ base,
                                                  int row, int rowmax, int ld, int k0, int lane) {
    if (row >= rowmax) { v16h z = {}; return z; }
    return load_tile16(base, row, ld, k0, lane);
}

__device__ __forceinline__ void store_tile_f32(float* __restrict__ C, int ld,
                                               int mbase, int n, const v8f& acc) {
#pragma unroll
    for (int j = 0; j < 8; ++j) C[(long)(mbase + j) * ld + n] = acc[j];
}

// ---------------- Stage 1: xz = x @ in_proj_w^T  (6272x384, K=96) ----------------
__global__ __launch_bounds__(32) void k_gemm_inproj(const float* __restrict__ X,
                                                    const float* __restrict__ Wt,
                                                    float* __restrict__ C) {
    int lane = threadIdx.x;
    int row = blockIdx.x * 16 + (lane & 15);
    int col = blockIdx.y * 16 + (lane & 15);
    v8f acc = {};
#pragma unroll
    for (int k0 = 0; k0 < DM; k0 += 32) {
        v16h a = load_tile16(X,  row, DM, k0, lane);
        v16h b = load_tile16(Wt, col, DM, k0, lane);
        acc = __builtin_amdgcn_wmma_f32_16x16x32_f16(false, a, false, b, (short)0, acc, false, false);
    }
    int n = blockIdx.y * 16 + (lane & 15);
    int mbase = blockIdx.x * 16 + ((lane >> 4) << 3);
    store_tile_f32(C, 2 * DI, mbase, n, acc);
}

// ---------------- Stage 2: depthwise 3x3 conv + bias + SiLU + hscan1 store ----------------
__global__ void k_conv_silu(const float* __restrict__ xz,
                            const float* __restrict__ cw,
                            const float* __restrict__ cb,
                            float* __restrict__ x1cl) {
    int idx = blockIdx.x * blockDim.x + threadIdx.x;
    if (idx >= BB * HH * WW * DI) return;
    int c = idx % DI;
    int w = (idx / DI) % WW;
    int h = (idx / (DI * WW)) % HH;
    int b = idx / (DI * WW * HH);
    float acc = cb[c];
#pragma unroll
    for (int kh = 0; kh < 3; ++kh) {
        int hy = h + kh - 1;
        if (hy < 0 || hy >= HH) continue;
#pragma unroll
        for (int kw = 0; kw < 3; ++kw) {
            int wx = w + kw - 1;
            if (wx < 0 || wx >= WW) continue;
            float xv = xz[(((long)b * LL + hy * WW + wx)) * (2 * DI) + c];
            acc += xv * cw[c * 9 + kh * 3 + kw];
        }
    }
    float s = acc / (1.f + __expf(-acc));           // SiLU
    int hdst = (w & 1) ? (HH - 1 - h) : h;          // hscan1 (involution)
    x1cl[((long)b * LL + hdst * WW + w) * DI + c] = s;
}

// ---------------- Stage 3: x_dbl = xs @ x_proj_weight^T  (per b,k: 3136x38, K=192) ----------------
__global__ __launch_bounds__(32) void k_gemm_xproj(const float* __restrict__ x1cl,
                                                   const float* __restrict__ xpw,
                                                   float* __restrict__ xdbl) {
    int lane = threadIdx.x;
    int z = blockIdx.z;           // b*KK + k
    int b = z >> 2, k = z & 3;
    const float* A = x1cl + (long)b * LL * DI;
    const float* Bm = xpw + (long)k * XPN * DI;
    int l   = blockIdx.x * 16 + (lane & 15);
    int col = blockIdx.y * 16 + (lane & 15);
    long arow = xs_src(k, l);
    v8f acc = {};
#pragma unroll
    for (int k0 = 0; k0 < DI; k0 += 32) {
        v16h a = load_tile16(A, arow, DI, k0, lane);
        v16h bt = load_tile16_guard(Bm, col, XPN, DI, k0, lane);
        acc = __builtin_amdgcn_wmma_f32_16x16x32_f16(false, a, false, bt, (short)0, acc, false, false);
    }
    int n = blockIdx.y * 16 + (lane & 15);
    if (n >= XPN) return;
    int mbase = blockIdx.x * 16 + ((lane >> 4) << 3);
    float* C = xdbl + (long)z * LL * XPNP;
#pragma unroll
    for (int j = 0; j < 8; ++j) C[(long)(mbase + j) * XPNP + n] = acc[j];
}

// ---------------- Stage 4: dts = x_dbl[:, :DTR] @ dt_projs_weight^T  (K=6, VALU) ----------------
__global__ void k_dtproj(const float* __restrict__ xdbl,
                         const float* __restrict__ dtw,
                         float* __restrict__ dts) {
    long idx = (long)blockIdx.x * blockDim.x + threadIdx.x;
    if (idx >= (long)BB * KK * LL * DI) return;
    int d = idx % DI;
    long bl = idx / DI;           // (b*KK+k)*LL + l
    int k = (bl / LL) & 3;
    const float* r = xdbl + bl * XPNP;
    const float* wv = dtw + ((long)k * DI + d) * DTR;
    float acc = 0.f;
#pragma unroll
    for (int q = 0; q < DTR; ++q) acc += r[q] * wv[q];
    dts[idx] = acc;
}

// ---------------- Stage 5a: chunk summaries (P = prod a, F = zero-init partial state) ----------
__global__ void k_scan_chunk(const float* __restrict__ x1cl,
                             const float* __restrict__ xdbl,
                             const float* __restrict__ dts,
                             const float* __restrict__ A_logs,
                             const float* __restrict__ dt_bias,
                             float* __restrict__ Pbuf,
                             float* __restrict__ Fbuf) {
    long t = (long)blockIdx.x * blockDim.x + threadIdx.x;
    if (t >= (long)ELEMS * NC) return;
    int e  = (int)(t % ELEMS);
    int c  = (int)(t / ELEMS);
    int n  = e % NS;
    int ch = e / NS;              // (b*KK+k)*DI + d
    int d  = ch % DI;
    int bk = ch / DI;
    int k  = bk & 3, b = bk >> 2;
    int kd = k * DI + d;

    float An   = -__expf(A_logs[(long)kd * NS + n]);
    float bias = dt_bias[kd];
    const float* dtp = dts  + (long)bk * LL * DI + d;
    const float* bcp = xdbl + (long)bk * LL * XPNP;
    const float* xb  = x1cl + (long)b * LL * DI + d;

    float P = 1.f, F = 0.f;
    int l0 = c * CS;
    for (int s = 0; s < CS; ++s) {
        int l = l0 + s;
        float sp = softplus_f(dtp[(long)l * DI] + bias);
        float u  = xb[(long)xs_src(k, l) * DI];
        float a  = __expf(sp * An);
        F = a * F + sp * bcp[(long)l * XPNP + DTR + n] * u;
        P *= a;
    }
    Pbuf[(long)c * ELEMS + e] = P;
    Fbuf[(long)c * ELEMS + e] = F;
}

// ---------------- Stage 5b: carry across chunks (serial over 28 chunks only) ----------------
__global__ void k_scan_carry(const float* __restrict__ Pbuf,
                             const float* __restrict__ Fbuf,
                             float* __restrict__ initbuf) {
    int e = blockIdx.x * blockDim.x + threadIdx.x;
    if (e >= ELEMS) return;
    float carry = 0.f;
    for (int c = 0; c < NC; ++c) {
        initbuf[(long)c * ELEMS + e] = carry;
        carry = Fbuf[(long)c * ELEMS + e] + Pbuf[(long)c * ELEMS + e] * carry;
    }
}

// ---------------- Stage 5c: replay chunk with correct init; y via 16-lane shfl tree ----------
// wave = 2 channels x 16 states; grid = (NCH/2, NC)
__global__ __launch_bounds__(32) void k_scan_out(const float* __restrict__ x1cl,
                                                 const float* __restrict__ xdbl,
                                                 const float* __restrict__ dts,
                                                 const float* __restrict__ A_logs,
                                                 const float* __restrict__ Ds,
                                                 const float* __restrict__ dt_bias,
                                                 const float* __restrict__ initbuf,
                                                 float* __restrict__ oy) {
    int lane = threadIdx.x;
    int n  = lane & 15;
    int ch = blockIdx.x * 2 + (lane >> 4);
    int c  = blockIdx.y;
    int d  = ch % DI;
    int bk = ch / DI;
    int k  = bk & 3, b = bk >> 2;
    int kd = k * DI + d;

    float An   = -__expf(A_logs[(long)kd * NS + n]);
    float bias = dt_bias[kd];
    float Dv   = Ds[kd];
    float h    = initbuf[(long)c * ELEMS + (long)ch * NS + n];
    const float* dtp = dts  + (long)bk * LL * DI + d;
    const float* bcp = xdbl + (long)bk * LL * XPNP;
    const float* xb  = x1cl + (long)b * LL * DI + d;
    float* yo = oy + (long)bk * LL * DI + d;

    int l0 = c * CS;
    for (int s = 0; s < CS; ++s) {
        int l = l0 + s;
        float sp = softplus_f(dtp[(long)l * DI] + bias);
        float u  = xb[(long)xs_src(k, l) * DI];
        float a  = __expf(sp * An);
        const float* bc = bcp + (long)l * XPNP;
        h = a * h + sp * bc[DTR + n] * u;
        float tsum = h * bc[DTR + NS + n];
        // reduce over the 16 states of this half-wave group (xor of bits 0..3 stays in group)
        tsum += __shfl_xor(tsum, 8, 32);
        tsum += __shfl_xor(tsum, 4, 32);
        tsum += __shfl_xor(tsum, 2, 32);
        tsum += __shfl_xor(tsum, 1, 32);
        if (n == 0) yo[(long)l * DI] = tsum + u * Dv;
    }
}

// ---------------- Stage 6: gather 4 directions + LayerNorm + *silu(z) ----------------
__global__ __launch_bounds__(32) void k_gather_ln(const float* __restrict__ oy,
                                                  const float* __restrict__ xz,
                                                  const float* __restrict__ nw,
                                                  const float* __restrict__ nb,
                                                  float* __restrict__ yln) {
    int bl = blockIdx.x;
    int l = bl % LL, b = bl / LL;
    int h = l / WW, w = l % WW;
    int hp = (w & 1) ? (HH - 1 - h) : h;
    int l0 = hp * WW + w;
    int i1 = (hp & 1) ? (WW - 1 - w) : w;
    int l1 = i1 * HH + hp;
    long base = (long)b * KK * LL * DI;
    const float* o0 = oy + base + ((long)0 * LL + l0) * DI;
    const float* o2 = oy + base + ((long)2 * LL + (LL - 1 - l0)) * DI;
    const float* o1 = oy + base + ((long)1 * LL + l1) * DI;
    const float* o3 = oy + base + ((long)3 * LL + (LL - 1 - l1)) * DI;

    int lane = threadIdx.x;
    float v[6];
    float s = 0.f, q = 0.f;
#pragma unroll
    for (int i = 0; i < 6; ++i) {
        int d = lane + 32 * i;
        float y = o0[d] + o1[d] + o2[d] + o3[d];
        v[i] = y; s += y; q += y * y;
    }
#pragma unroll
    for (int off = 16; off; off >>= 1) {        // wave32 reduction
        s += __shfl_xor(s, off, 32);
        q += __shfl_xor(q, off, 32);
    }
    float mu = s * (1.f / DI);
    float var = q * (1.f / DI) - mu * mu;
    float rstd = rsqrtf(var + 1e-5f);
    const float* zrow = xz + (long)bl * (2 * DI) + DI;
    float* out = yln + (long)bl * DI;
#pragma unroll
    for (int i = 0; i < 6; ++i) {
        int d = lane + 32 * i;
        float zv = zrow[d];
        float sz = zv / (1.f + __expf(-zv));
        out[d] = ((v[i] - mu) * rstd * nw[d] + nb[d]) * sz;
    }
}

// ---------------- Stage 7: out = yln @ out_proj_w^T  (6272x96, K=192) ----------------
__global__ __launch_bounds__(32) void k_gemm_outproj(const float* __restrict__ Y,
                                                     const float* __restrict__ Wt,
                                                     float* __restrict__ C) {
    int lane = threadIdx.x;
    int row = blockIdx.x * 16 + (lane & 15);
    int col = blockIdx.y * 16 + (lane & 15);
    v8f acc = {};
#pragma unroll
    for (int k0 = 0; k0 < DI; k0 += 32) {
        v16h a = load_tile16(Y,  row, DI, k0, lane);
        v16h b = load_tile16(Wt, col, DI, k0, lane);
        acc = __builtin_amdgcn_wmma_f32_16x16x32_f16(false, a, false, b, (short)0, acc, false, false);
    }
    int n = blockIdx.y * 16 + (lane & 15);
    int mbase = blockIdx.x * 16 + ((lane >> 4) << 3);
    store_tile_f32(C, DM, mbase, n, acc);
}

// ---------------- launch ----------------
extern "C" void kernel_launch(void* const* d_in, const int* in_sizes, int n_in,
                              void* d_out, int out_size, void* d_ws, size_t ws_size,
                              hipStream_t stream) {
    const float* x        = (const float*)d_in[0];
    const float* inpw     = (const float*)d_in[1];
    const float* convw    = (const float*)d_in[2];
    const float* convb    = (const float*)d_in[3];
    const float* xpw      = (const float*)d_in[4];
    const float* dtw      = (const float*)d_in[5];
    const float* dtb      = (const float*)d_in[6];
    const float* Alogs    = (const float*)d_in[7];
    const float* Ds       = (const float*)d_in[8];
    const float* normw    = (const float*)d_in[9];
    const float* normb    = (const float*)d_in[10];
    const float* outw     = (const float*)d_in[11];
    float* out = (float*)d_out;

    float* ws = (float*)d_ws;
    const long SZ_XZ   = (long)BB * LL * 2 * DI;     // 2,408,448 floats
    const long SZ_X1   = (long)BB * LL * DI;         // 1,204,224
    const long SZ_XDBL = (long)BB * KK * LL * XPNP;  // 1,204,224
    const long SZ_DTS  = (long)BB * KK * LL * DI;    // 4,816,896
    const long SZ_PF   = (long)NC * ELEMS;           //   688,128
    float* xz    = ws;
    float* x1cl  = xz    + SZ_XZ;
    float* xdbl  = x1cl  + SZ_X1;
    float* dts   = xdbl  + SZ_XDBL;
    float* oy    = dts   + SZ_DTS;
    float* yln   = oy    + SZ_DTS;
    float* Pbuf  = yln   + SZ_X1;
    float* Fbuf  = Pbuf  + SZ_PF;
    float* initb = Fbuf  + SZ_PF;
    (void)ws_size; (void)in_sizes; (void)n_in; (void)out_size;

    // 1) in_proj GEMM: (B*L, 384) = (B*L, 96) x (384, 96)^T
    k_gemm_inproj<<<dim3((BB * LL) / 16, (2 * DI) / 16), 32, 0, stream>>>(x, inpw, xz);

    // 2) depthwise conv + SiLU + hscan1
    {
        long n = (long)BB * HH * WW * DI;
        k_conv_silu<<<(n + 255) / 256, 256, 0, stream>>>(xz, convw, convb, x1cl);
    }

    // 3) x_proj GEMM per (b,k): (L, 38) = xs(L,192) x W_k(38,192)^T
    k_gemm_xproj<<<dim3(LL / 16, XPNP / 16, BB * KK), 32, 0, stream>>>(x1cl, xpw, xdbl);

    // 4) dt projection (K=6, VALU)
    {
        long n = (long)BB * KK * LL * DI;
        k_dtproj<<<(n + 255) / 256, 256, 0, stream>>>(xdbl, dtw, dts);
    }

    // 5) chunked selective scan: summaries -> carries -> replay with output
    {
        long n = (long)ELEMS * NC;
        k_scan_chunk<<<(n + 255) / 256, 256, 0, stream>>>(x1cl, xdbl, dts, Alogs, dtb, Pbuf, Fbuf);
        k_scan_carry<<<(ELEMS + 255) / 256, 256, 0, stream>>>(Pbuf, Fbuf, initb);
        k_scan_out<<<dim3(NCH / 2, NC), 32, 0, stream>>>(x1cl, xdbl, dts, Alogs, Ds, dtb, initb, oy);
    }

    // 6) gather directions + LayerNorm + silu(z)
    k_gather_ln<<<BB * LL, 32, 0, stream>>>(oy, xz, normw, normb, yln);

    // 7) out_proj GEMM: (B*L, 96) = (B*L, 192) x (96, 192)^T
    k_gemm_outproj<<<dim3((BB * LL) / 16, DM / 16), 32, 0, stream>>>(yln, outw, out);
}
